// CostVolumeBuilder_24713241821735
// MI455X (gfx1250) — compile-verified, matrix operand order checked by reading, not run on hardware
//
#include <hip/hip_runtime.h>

// ---------------------------------------------------------------------------
// Cost volume (correlation layer) for MI455X / gfx1250.
//   out[b, (dy+4)*9+(dx+4), h, w] = (1/sqrt(C)) * sum_c f1[b,c,h,w]*f2[b,c,h+dy,w+dx]
// B=4, C=128, H=192, W=256, 81 displacements. fp32 in / fp32 out.
//
// Per wave: 16-pixel w-tile forms A (16x128, bf16). Per dy, feat2 columns
// [w0-4 .. w0+27] form two 16x16 GEMM tiles; 8x v_wmma_f32_16x16x32_bf16
// per dy, then scatter the 9-wide diagonal band (dx index = j - m).
// feat2's 9x dy-reuse is L2-resident (66 MB working set / batch vs 192 MB L2).
// bf16 packing: v_add(+0x8000) + v_perm_b32 (3 VALU per pair); padding guards
// are scalar-branched (readfirstlane) so the hot path has no cndmasks and
// EXEC stays all-ones for every WMMA.
// ---------------------------------------------------------------------------

typedef __attribute__((ext_vector_type(16))) __bf16        v16bf;
typedef __attribute__((ext_vector_type(8)))  float         v8f;
typedef __attribute__((ext_vector_type(8)))  unsigned int  v8u;

#define BB    4
#define CC    128
#define HH    192
#define WW    256
#define HWSZ  (HH * WW)
#define DMAX  4
#define NDISP 81
#define WTILES (WW / 16)          // 16 tiles per row
#define NTILES (BB * HH * WTILES) // 12288 waves total

// fp32 -> bf16 (round-half-up) for two floats, packed into one dword.
// v_perm_b32 sel 0x07060302 picks {uh[31:16], ul[31:16]}.
__device__ __forceinline__ unsigned int bf16pk(float lo, float hi) {
    const unsigned int ul = __float_as_uint(lo) + 0x8000u;
    const unsigned int uh = __float_as_uint(hi) + 0x8000u;
    return __builtin_amdgcn_perm(uh, ul, 0x07060302u);
}

// One 16x32 bf16 WMMA operand fragment, CDNA5 per-lane layout:
//   lanes 0-15 : K = khalf + {0..7} in v0..v3, khalf + {16..23} in v4..v7
//   lanes 16-31: same, khalf = 8.
// p points at (k=0) for this lane, channel stride HWSZ, always in-bounds.
template <bool GUARD>
__device__ __forceinline__ v16bf load_frag(const float* __restrict__ p,
                                           int khalf, bool valid) {
    v8u pk;
#pragma unroll
    for (int i = 0; i < 8; ++i) {
        const int k0 = ((i >= 4) ? 16 : 0) + khalf + 2 * (i & 3);
        const float f0 = p[(size_t)k0 * HWSZ];
        const float f1 = p[(size_t)(k0 + 1) * HWSZ];
        const unsigned int v = bf16pk(f0, f1);
        pk[i] = GUARD ? (valid ? v : 0u) : v;
    }
    return __builtin_bit_cast(v16bf, pk);
}

__device__ __forceinline__ v8f wmma_bf16(v16bf a, v16bf b, v8f c) {
    return __builtin_amdgcn_wmma_f32_16x16x32_bf16(
        false, a, false, b, (short)0, c, false, false);
}

__global__ __launch_bounds__(256)
void cost_volume_wmma_bf16(const float* __restrict__ feat1,
                           const float* __restrict__ feat2,
                           float* __restrict__ out) {
    const int lane = threadIdx.x & 31;
    const int wv   = threadIdx.x >> 5;
    const int tile = blockIdx.x * 8 + wv;         // grid sized exactly: no mask

    const int wt = tile & (WTILES - 1);
    const int h  = (tile / WTILES) % HH;
    const int b  = tile / (WTILES * HH);
    const int w0 = wt << 4;

    const int n     = lane & 15;        // A row (pixel) / B column / C column
    const int khalf = (lane >> 4) << 3; // 0 or 8
    const int mhigh = khalf;            // C/D: VGPR r holds M = r + mhigh

    const float* f1b = feat1 + (size_t)b * CC * HWSZ;
    const float* f2b = feat2 + (size_t)b * CC * HWSZ;

    // ---- A: feat1 tile, always interior, reused for all 9 dy --------------
    const float* a_base = f1b + (size_t)h * WW + (w0 + n);
    v16bf afr[4];
#pragma unroll
    for (int kf = 0; kf < 4; ++kf)
        afr[kf] = load_frag<false>(a_base + (size_t)kf * 32 * HWSZ, khalf, true);

    const float  scale = 0.08838834764831845f;    // 1/sqrt(128)
    const size_t obase = (size_t)b * NDISP * HWSZ + (size_t)h * WW + w0;
    const bool   col_edge = (w0 == 0) || (w0 == WW - 16);   // wave-uniform

#pragma unroll 1
    for (int dy = -DMAX; dy <= DMAX; ++dy) {
        const int  dyi   = dy + DMAX;
        const int  h2    = h + dy;
        const bool rowok = (h2 >= 0) && (h2 < HH);          // wave-uniform

        if (!__builtin_amdgcn_readfirstlane((int)rowok)) {
            // whole dy band is zero padding: store zeros, skip loads + WMMA
#pragma unroll
            for (int r = 0; r < 8; ++r) {
                const int m = r + mhigh;
                const int dxi0 = n - m;
                if (dxi0 >= 0 && dxi0 <= 8)
                    out[obase + (size_t)(dyi * 9 + dxi0) * HWSZ + m] = 0.f;
                const int dxi1 = n + 16 - m;
                if (dxi1 >= 0 && dxi1 <= 8)
                    out[obase + (size_t)(dyi * 9 + dxi1) * HWSZ + m] = 0.f;
            }
            continue;
        }

        const float* b_row = f2b + (size_t)h2 * WW;
        if (dy < DMAX)
            __builtin_prefetch(b_row + WW + w0, 0, 1);  // global_prefetch_b8

        const int p0 = w0 - 4 + n;                // B tile 0: band cols j = 0..15
        const int p1 = w0 + 12 + n;               // B tile 1: band cols j = 16..31

        v8f acc0 = {}; v8f acc1 = {};
        if (!__builtin_amdgcn_readfirstlane((int)col_edge)) {
            // fast path: 16 <= w0 <= 224 -> p0 in [12,251], p1 in [28,251]
#pragma unroll
            for (int kf = 0; kf < 4; ++kf) {
                const float* bk = b_row + (size_t)kf * 32 * HWSZ;
                v16bf bf0 = load_frag<false>(bk + p0, khalf, true);
                v16bf bf1 = load_frag<false>(bk + p1, khalf, true);
                acc0 = wmma_bf16(afr[kf], bf0, acc0);
                acc1 = wmma_bf16(afr[kf], bf1, acc1);
            }
        } else {
            // edge w-tiles: clamp addresses, zero-fill out-of-image columns
            const bool ok0 = (p0 >= 0) && (p0 < WW);
            const bool ok1 = (p1 < WW);
            const int  p0c = ok0 ? p0 : 0;
            const int  p1c = ok1 ? p1 : 0;
#pragma unroll
            for (int kf = 0; kf < 4; ++kf) {
                const float* bk = b_row + (size_t)kf * 32 * HWSZ;
                v16bf bf0 = load_frag<true>(bk + p0c, khalf, ok0);
                v16bf bf1 = load_frag<true>(bk + p1c, khalf, ok1);
                acc0 = wmma_bf16(afr[kf], bf0, acc0);
                acc1 = wmma_bf16(afr[kf], bf1, acc1);
            }
        }

        // ---- scatter the 9-wide band: dx index = j - m, valid when 0..8 ---
#pragma unroll
        for (int r = 0; r < 8; ++r) {
            const int m = r + mhigh;              // output pixel = w0 + m
            const int dxi0 = n - m;               // tile 0: j = n
            if (dxi0 >= 0 && dxi0 <= 8)
                out[obase + (size_t)(dyi * 9 + dxi0) * HWSZ + m] = acc0[r] * scale;
            const int dxi1 = n + 16 - m;          // tile 1: j = n + 16
            if (dxi1 >= 0 && dxi1 <= 8)
                out[obase + (size_t)(dyi * 9 + dxi1) * HWSZ + m] = acc1[r] * scale;
        }
    }
}

extern "C" void kernel_launch(void* const* d_in, const int* in_sizes, int n_in,
                              void* d_out, int out_size, void* d_ws, size_t ws_size,
                              hipStream_t stream) {
    const float* feat1 = (const float*)d_in[0];
    const float* feat2 = (const float*)d_in[1];
    float*       out   = (float*)d_out;

    // 12288 tiles, 8 waves (256 threads) per block -> 1536 blocks.
    // Block order: w-tiles fastest, then h, then b => L2-coherent feat2 reuse.
    dim3 block(256);
    dim3 grid(NTILES / 8);
    cost_volume_wmma_bf16<<<grid, block, 0, stream>>>(feat1, feat2, out);
}
